// MoEMLP_3762391351684
// MI455X (gfx1250) — matrix operand order repaired and checked by reading, not compiled, not run on hardware
//
#include <hip/hip_runtime.h>
#include <hip/hip_bf16.h>
#include <cstdint>

typedef __attribute__((ext_vector_type(16))) __bf16 v16bf;
typedef __attribute__((ext_vector_type(8)))  float  v8f;
typedef __attribute__((ext_vector_type(4)))  unsigned u32x4;
typedef __attribute__((ext_vector_type(8)))  int      i32x8;
typedef __attribute__((ext_vector_type(4)))  int      i32x4;
typedef __attribute__((address_space(3)))    __bf16   lds_bf16_t;

#define T_TOK 2048
#define DIM   1024
#define NEXP  16
#define TOPK  4
#define IDIM  512

#define TILE_T 64
#define XS_STRIDE (DIM + 8)    // bf16 elems; pad so 16 rows hit distinct LDS banks
#define HS_STRIDE (IDIM + 8)   // matches TDM pad: 4 DWORDs after every 256 DWORDs

union FragA { v16bf v; float4 q[2]; };
union F16f  { float4 q[4]; float f[16]; };
union Pack4 { ushort4 u; __bf16 b[4]; };

// ---------------------------------------------------------------- init tail
__global__ void moe_init_tail(float* __restrict__ tail) {
    if (threadIdx.x < NEXP + 1) tail[threadIdx.x] = 0.0f;
}

// ---------------------------------------------------------------- router + top-k
__global__ __launch_bounds__(256) void moe_router(
    const float* __restrict__ x, const float* __restrict__ router,
    const float* __restrict__ rbias, float* __restrict__ combine,
    float* __restrict__ counts) {
    __shared__ float rs[256 * NEXP];
    __shared__ int   cnt[NEXP];
    const int tid = threadIdx.x;
    const int t   = blockIdx.x * 256 + tid;
    if (tid < NEXP) cnt[tid] = 0;

    float acc[NEXP];
#pragma unroll
    for (int e = 0; e < NEXP; ++e) acc[e] = 0.0f;

    for (int c0 = 0; c0 < DIM; c0 += 256) {
        __syncthreads();
        const float4* src = reinterpret_cast<const float4*>(router + (size_t)(c0 + tid) * NEXP);
        float4* dst = reinterpret_cast<float4*>(rs + tid * NEXP);
#pragma unroll
        for (int q = 0; q < 4; ++q) dst[q] = src[q];
        __syncthreads();
#pragma unroll 4
        for (int d = 0; d < 256; d += 4) {
            float4 xv = *reinterpret_cast<const float4*>(x + (size_t)t * DIM + c0 + d);
            const float* r0 = rs + d * NEXP;
#pragma unroll
            for (int e = 0; e < NEXP; ++e)
                acc[e] += xv.x * r0[e] + xv.y * r0[NEXP + e] +
                          xv.z * r0[2 * NEXP + e] + xv.w * r0[3 * NEXP + e];
        }
    }
    __syncthreads();

    float biased[NEXP];
#pragma unroll
    for (int e = 0; e < NEXP; ++e) biased[e] = acc[e] + rbias[e];

    int   sel[TOPK];
    float sw[TOPK];
    float ssum = 0.0f;
    unsigned used = 0;
#pragma unroll
    for (int k = 0; k < TOPK; ++k) {
        float best = -3.4e38f; int bi = 0;
#pragma unroll
        for (int e = 0; e < NEXP; ++e)
            if (!((used >> e) & 1u) && biased[e] > best) { best = biased[e]; bi = e; }
        used |= 1u << bi;
        sel[k] = bi;
        float s = 1.0f / (1.0f + __expf(-acc[bi]));
        sw[k] = s; ssum += s;
    }
    float cw[NEXP];
#pragma unroll
    for (int e = 0; e < NEXP; ++e) cw[e] = 0.0f;
#pragma unroll
    for (int k = 0; k < TOPK; ++k) {
        cw[sel[k]] += sw[k] / ssum;
        atomicAdd(&cnt[sel[k]], 1);
    }
#pragma unroll
    for (int e = 0; e < NEXP; e += 4)
        *reinterpret_cast<float4*>(combine + (size_t)t * NEXP + e) =
            make_float4(cw[e], cw[e + 1], cw[e + 2], cw[e + 3]);

    __syncthreads();
    if (tid < NEXP) atomicAdd(&counts[tid], (float)cnt[tid]);
}

// ---------------------------------------------------------------- entropy
__global__ void moe_entropy(const float* __restrict__ counts, float* __restrict__ entropy) {
    if (threadIdx.x == 0) {
        float total = 0.0f;
        for (int e = 0; e < NEXP; ++e) total += counts[e];
        total = fmaxf(total, 1.0f);
        float ent = 0.0f;
        for (int e = 0; e < NEXP; ++e) {
            float f = counts[e] / total;
            ent -= f * logf(f + 1e-6f);
        }
        *entropy = ent;
    }
}

// ---------------------------------------------------------------- gate/up GEMM + SiLU
__global__ __launch_bounds__(256) void moe_gate_up(
    const float* __restrict__ x, const float* __restrict__ wgu,
    const float* __restrict__ combine, __bf16* __restrict__ h) {
    __shared__ __bf16 Xs[TILE_T * XS_STRIDE];
    __shared__ float  c_s[TILE_T];
    const int tid  = threadIdx.x;
    const int lane = tid & 31;
    const int wave = tid >> 5;
    const int tile = blockIdx.x;
    const int e    = blockIdx.y;
    const int t0   = tile * TILE_T;

    // Stage X tile (64 x 1024) fp32 -> bf16 into LDS.
    for (int q = tid; q < TILE_T * DIM / 4; q += 256) {
        int row = q >> 8;
        int col = (q & 255) * 4;
        float4 v = *reinterpret_cast<const float4*>(x + (size_t)(t0 + row) * DIM + col);
        Pack4 p;
        p.b[0] = (__bf16)v.x; p.b[1] = (__bf16)v.y;
        p.b[2] = (__bf16)v.z; p.b[3] = (__bf16)v.w;
        *reinterpret_cast<ushort4*>(&Xs[row * XS_STRIDE + col]) = p.u;
    }
    if (tid < TILE_T) c_s[tid] = combine[(size_t)(t0 + tid) * NEXP + e];
    __syncthreads();

    const int mtile = wave >> 1;                 // 4 token sub-tiles
    const int jhalf = wave & 1;                  // 2 waves split the 32 I-groups
    const int mrow  = mtile * 16 + (lane & 15);
    const int koff  = (lane < 16) ? 0 : 8;       // 16-bit A layout (ISA 7.12.2)

    for (int jg = jhalf * 16; jg < jhalf * 16 + 16; ++jg) {
        const int jcol = jg * 16;
        v8f accg0 = {}, accg1 = {};
        v8f accu0 = {}, accu1 = {};
        for (int k0 = 0; k0 < DIM; k0 += 64) {
            FragA a0, a1;
            a0.q[0] = *reinterpret_cast<const float4*>(&Xs[mrow * XS_STRIDE + k0 + koff]);
            a0.q[1] = *reinterpret_cast<const float4*>(&Xs[mrow * XS_STRIDE + k0 + koff + 16]);
            a1.q[0] = *reinterpret_cast<const float4*>(&Xs[mrow * XS_STRIDE + k0 + 32 + koff]);
            a1.q[1] = *reinterpret_cast<const float4*>(&Xs[mrow * XS_STRIDE + k0 + 32 + koff + 16]);
            const float* wr0 = wgu + ((size_t)e * DIM + k0 + koff) * (2 * IDIM) + jcol + (lane & 15);
            const float* wr1 = wr0 + (size_t)32 * (2 * IDIM);
            __builtin_prefetch(wr0 + (size_t)64 * 2 * IDIM, 0, 1);
            F16f wg0, wu0, wg1, wu1;
#pragma unroll
            for (int q = 0; q < 16; ++q) {
                const size_t ko = (size_t)(q < 8 ? q : q + 8) * (2 * IDIM);
                wg0.f[q] = wr0[ko];
                wu0.f[q] = wr0[ko + IDIM];
                wg1.f[q] = wr1[ko];
                wu1.f[q] = wr1[ko + IDIM];
            }
            v16bf bg0, bu0, bg1, bu1;
#pragma unroll
            for (int i = 0; i < 16; ++i) {
                bg0[i] = (__bf16)wg0.f[i]; bu0[i] = (__bf16)wu0.f[i];
                bg1[i] = (__bf16)wg1.f[i]; bu1[i] = (__bf16)wu1.f[i];
            }
            accg0 = __builtin_amdgcn_wmma_f32_16x16x32_bf16(false, a0.v, false, bg0,
                                                            (short)0, accg0, false, false);
            accu0 = __builtin_amdgcn_wmma_f32_16x16x32_bf16(false, a0.v, false, bu0,
                                                            (short)0, accu0, false, false);
            accg1 = __builtin_amdgcn_wmma_f32_16x16x32_bf16(false, a1.v, false, bg1,
                                                            (short)0, accg1, false, false);
            accu1 = __builtin_amdgcn_wmma_f32_16x16x32_bf16(false, a1.v, false, bu1,
                                                            (short)0, accu1, false, false);
        }
        v8f accg = accg0 + accg1;
        v8f accu = accu0 + accu1;
#pragma unroll
        for (int r = 0; r < 8; ++r) {
            int mloc = mtile * 16 + r + ((lane < 16) ? 0 : 8);
            float g = accg[r], u = accu[r];
            float sig  = __builtin_amdgcn_rcpf(1.0f + __expf(-g));   // fast sigmoid
            float hval = g * sig * u * c_s[mloc];
            int t = t0 + mloc;
            int i = jcol + (lane & 15);
            h[((size_t)e * T_TOK + t) * IDIM + i] = (__bf16)hval;
        }
    }
}

// ---------------------------------------------------------------- TDM issue helper
// DMA a 64x512 bf16 tile (rows contiguous, row stride 512 elems) into LDS at
// byte offset `lds_addr`, hardware-padding +4 DWORDs (8 bf16) after each
// 256-DWORD row so the LDS image has HS_STRIDE layout.
__device__ __forceinline__ void tdm_load_h_tile(unsigned lds_addr, const __bf16* gsrc) {
    unsigned long long ga = (unsigned long long)(uintptr_t)gsrc;
    u32x4 g0;
    g0[0] = 1u;                                   // count=1 (valid descriptor)
    g0[1] = lds_addr;                             // lds_addr
    g0[2] = (unsigned)(ga & 0xFFFFFFFFu);         // global_addr[31:0]
    g0[3] = (unsigned)((ga >> 32) & 0x01FFFFFFu)  // global_addr[56:32]
            | (2u << 30);                         // type=2 (image)
    i32x8 g1;
    g1[0] = (1 << 16)      // data_size = 2 bytes
          | (1 << 20)      // pad_enable
          | (7 << 22)      // pad_interval = 256 DWORDs (one 512-elem row)
          | (3 << 25);     // pad_amount  = 4 DWORDs (8 bf16)
    g1[1] = (int)(512u << 16);   // tensor_dim0 = 512 (low 16)
    g1[2] = (int)(64u << 16);    // tensor_dim0 hi=0 | tensor_dim1 = 64 (low 16)
    g1[3] = (int)(512u << 16);   // tensor_dim1 hi=0 | tile_dim0 = 512
    g1[4] = 64;                  // tile_dim1 = 64 | tile_dim2 = 0
    g1[5] = 512;                 // tensor_dim0_stride[31:0] = 512 elems
    g1[6] = 0;
    g1[7] = 0;
    i32x4 gz  = {0, 0, 0, 0};                  // 2D tensor: groups 2/3 unused
    i32x8 gz8 = {0, 0, 0, 0, 0, 0, 0, 0};      // extra group (clang-23 6-arg form)
    __builtin_amdgcn_tensor_load_to_lds(g0, g1, gz, gz, gz8, 0);
}

// ---------------------------------------------------------------- down GEMM (sum over experts)
__global__ __launch_bounds__(256) void moe_down(
    const __bf16* __restrict__ h, const float* __restrict__ wd,
    float* __restrict__ y) {
    __shared__ __bf16 Hs[2][TILE_T * HS_STRIDE];   // double-buffered TDM target
    const int tid  = threadIdx.x;
    const int lane = tid & 31;
    const int wave = tid >> 5;
    const int tile = blockIdx.x;
    const int d0   = blockIdx.y * 128;
    const int t0   = tile * TILE_T;

    const int mtile = wave & 3;
    const int nhalf = wave >> 2;
    const int mrow  = mtile * 16 + (lane & 15);
    const int koff  = (lane < 16) ? 0 : 8;

    // LDS byte offset of Hs for the TDM descriptors.
    const unsigned ldsBase  = (unsigned)(unsigned long long)(lds_bf16_t*)&Hs[0][0];
    const unsigned bufBytes = TILE_T * HS_STRIDE * 2u;

    v8f acc[4] = {};

    // Prime the pipeline: DMA expert 0's tile into buffer 0.
    if (wave == 0)
        tdm_load_h_tile(ldsBase, h + (size_t)t0 * IDIM);

    for (int e = 0; e < NEXP; ++e) {
        const int cur = e & 1;
        if (wave == 0) {
            if (e + 1 < NEXP) {
                // Prefetch next expert's tile into the other buffer; TDM ops
                // complete in order, so tensorcnt<=1 means load(e) has landed.
                tdm_load_h_tile(ldsBase + (unsigned)(cur ^ 1) * bufBytes,
                                h + ((size_t)(e + 1) * T_TOK + t0) * IDIM);
                __builtin_amdgcn_s_wait_tensorcnt(1);
            } else {
                __builtin_amdgcn_s_wait_tensorcnt(0);
            }
        }
        __syncthreads();   // load(e) complete & visible to all 8 waves

        const __bf16* hs = &Hs[cur][0];
        for (int i0 = 0; i0 < IDIM; i0 += 32) {
            FragA a;
            a.q[0] = *reinterpret_cast<const float4*>(&hs[mrow * HS_STRIDE + i0 + koff]);
            a.q[1] = *reinterpret_cast<const float4*>(&hs[mrow * HS_STRIDE + i0 + koff + 16]);
            const float* wr = wd + ((size_t)e * IDIM + i0 + koff) * DIM + d0 + nhalf * 64 + (lane & 15);
            __builtin_prefetch(wr + (size_t)32 * DIM, 0, 1);
#pragma unroll
            for (int n = 0; n < 4; ++n) {
                F16f wv;
#pragma unroll
                for (int q = 0; q < 16; ++q)
                    wv.f[q] = wr[(size_t)(q < 8 ? q : q + 8) * DIM + n * 16];
                v16bf b;
#pragma unroll
                for (int i = 0; i < 16; ++i) b[i] = (__bf16)wv.f[i];
                acc[n] = __builtin_amdgcn_wmma_f32_16x16x32_bf16(false, a.v, false, b,
                                                                 (short)0, acc[n], false, false);
            }
        }
        __syncthreads();   // all waves done with Hs[cur] before it is refilled at e+2
    }
#pragma unroll
    for (int n = 0; n < 4; ++n) {
#pragma unroll
        for (int r = 0; r < 8; ++r) {
            int mloc = mtile * 16 + r + ((lane < 16) ? 0 : 8);
            int col  = d0 + nhalf * 64 + n * 16 + (lane & 15);
            y[(size_t)(t0 + mloc) * DIM + col] = acc[n][r];
        }
    }
}

// ---------------------------------------------------------------- launch
extern "C" void kernel_launch(void* const* d_in, const int* in_sizes, int n_in,
                              void* d_out, int out_size, void* d_ws, size_t ws_size,
                              hipStream_t stream) {
    (void)in_sizes; (void)n_in; (void)out_size; (void)ws_size;
    const float* x      = (const float*)d_in[0];
    const float* router = (const float*)d_in[1];
    const float* rbias  = (const float*)d_in[2];
    const float* wgu    = (const float*)d_in[3];
    const float* wd     = (const float*)d_in[4];

    float* out     = (float*)d_out;
    float* counts  = out + (size_t)T_TOK * DIM;
    float* entropy = counts + NEXP;

    float*  combine = (float*)d_ws;
    __bf16* hws     = (__bf16*)((char*)d_ws + (size_t)T_TOK * NEXP * sizeof(float));

    moe_init_tail<<<1, 32, 0, stream>>>(counts);
    moe_router<<<T_TOK / 256, 256, 0, stream>>>(x, router, rbias, combine, counts);
    moe_entropy<<<1, 32, 0, stream>>>(counts, entropy);
    moe_gate_up<<<dim3(T_TOK / TILE_T, NEXP), 256, 0, stream>>>(x, wgu, combine, hws);
    moe_down<<<dim3(T_TOK / TILE_T, DIM / 128), 256, 0, stream>>>(hws, wd, out);
}
